// AttentionPooling_23811298690072
// MI455X (gfx1250) — compile-verified, matrix operand order checked by reading, not run on hardware
//
#include <hip/hip_runtime.h>
#include <hip/hip_bf16.h>
#include <stdint.h>

// ---- problem constants (match reference) ----
#define MOL_C   128
#define HID_C   256
#define CAT_C   384            // MOL_C + HID_C
#define TILE_M  64             // rows per workgroup (4 strips of 16)
#define NS      4              // M strips per wave
#define A_STRIDE 392           // bf16 elems per LDS row of A (384 + 8 pad -> conflict-free)
#define G_STRIDE 260           // f32 elems per LDS row of g  (256 + 4 pad)

typedef __attribute__((ext_vector_type(16))) __bf16        v16bf;
typedef __attribute__((ext_vector_type(8)))  float         v8f;
typedef __attribute__((ext_vector_type(8)))  unsigned int  u32x8;

// Single-instruction bf16 pair pack: {hi[31:16], lo[31:16]} via v_perm_b32.
__device__ __forceinline__ unsigned int pack_bf16(float lo, float hi) {
    return __builtin_amdgcn_perm(__builtin_bit_cast(unsigned int, hi),
                                 __builtin_bit_cast(unsigned int, lo),
                                 0x07060302u);
}

// Assemble a 16-bf16 wmma operand from two 16-byte chunks at p and p+32.
// (CDNA5 16-bit operand layout: a lane holds K sub-chunks [k, k+8) and [k+16, k+24),
//  with 'half' = lane>>4 shifting both chunks by 8.)
__device__ __forceinline__ v16bf load_bf16_pair(const char* p) {
    uint4 a0 = *(const uint4*)(p);
    uint4 a1 = *(const uint4*)(p + 32);
    u32x8 u = {a0.x, a0.y, a0.z, a0.w, a1.x, a1.y, a1.z, a1.w};
    return __builtin_bit_cast(v16bf, u);
}

// A-tile from the LDS-staged bf16 concat tile.
__device__ __forceinline__ v16bf load_a_tile(const char* sA, int rowA, int k, int half) {
    return load_bf16_pair(sA + ((unsigned)rowA * A_STRIDE + (unsigned)(k + half * 8)) * 2u);
}

// B-tile from pre-converted bf16 weights (row-major, row stride = rowlen elems).
__device__ __forceinline__ v16bf load_b_tile_bf(const unsigned short* wrow, int k, int half) {
    return load_bf16_pair((const char*)(wrow + k + half * 8));
}

// B-tile from f32 weights with inline conversion (fallback when d_ws too small).
__device__ __forceinline__ v16bf load_b_tile_f32(const float* __restrict__ wrow, int k, int half) {
    const float* wp = wrow + k + half * 8;
    float4 c0 = *(const float4*)(wp);
    float4 c1 = *(const float4*)(wp + 4);
    float4 c2 = *(const float4*)(wp + 16);
    float4 c3 = *(const float4*)(wp + 20);
    u32x8 u;
    u[0] = pack_bf16(c0.x, c0.y); u[1] = pack_bf16(c0.z, c0.w);
    u[2] = pack_bf16(c1.x, c1.y); u[3] = pack_bf16(c1.z, c1.w);
    u[4] = pack_bf16(c2.x, c2.y); u[5] = pack_bf16(c2.z, c2.w);
    u[6] = pack_bf16(c3.x, c3.y); u[7] = pack_bf16(c3.z, c3.w);
    return __builtin_bit_cast(v16bf, u);
}

#define WMMA_BF16(a, b, c) \
    __builtin_amdgcn_wmma_f32_16x16x32_bf16(false, (a), false, (b), (short)0, (c), false, false)

__global__ void __launch_bounds__(256) zero_out_kernel(float* __restrict__ p, int n) {
    int i = blockIdx.x * blockDim.x + threadIdx.x;
    if (i < n) p[i] = 0.0f;
}

// One-shot weight conversion: [W_lin | W_last] f32 -> packed bf16 in d_ws.
__global__ void __launch_bounds__(256)
convert_w_kernel(const float* __restrict__ wa, int na,   // element counts (even)
                 const float* __restrict__ wb, int nb,
                 unsigned int* __restrict__ dst)
{
    int i = blockIdx.x * blockDim.x + threadIdx.x;       // bf16-pair index
    int npa = na >> 1;
    int tot = (na + nb) >> 1;
    if (i >= tot) return;
    const float* src = (i < npa) ? wa : wb;
    int j = (i < npa) ? (i * 2) : ((i - npa) * 2);
    dst[i] = pack_bf16(src[j], src[j + 1]);
}

template <bool WBF>
__global__ void __launch_bounds__(256)
attn_pool_kernel(const float* __restrict__ input_rep,
                 const float* __restrict__ final_rep,
                 const int*   __restrict__ graph_index,
                 const float* __restrict__ W_lin,
                 const float* __restrict__ b_lin,
                 const float* __restrict__ W_last,
                 const float* __restrict__ b_last,
                 const unsigned short* __restrict__ wbf,  // bf16 [W_lin | W_last], or null
                 float*       __restrict__ out,
                 int n_rows)
{
    // Phase A: bf16 concat tile (64 x 392 x 2B = 49 KB)
    // Phase B: f32 gated tile   (64 x 260 x 4B = 65 KB)  -- same buffer, reused
    __shared__ __align__(16) char smem[TILE_M * G_STRIDE * 4];
    __shared__ int s_gi[TILE_M];

    const int tid  = threadIdx.x;
    const int wave = tid >> 5;
    const int lane = tid & 31;
    const int nlo  = lane & 15;
    const int half = lane >> 4;
    const int row0 = blockIdx.x * TILE_M;

    // ---- stage graph indices for this row block ----
    if (tid < TILE_M) {
        int r = row0 + tid;
        if (r >= n_rows) r = n_rows - 1;
        s_gi[tid] = graph_index[r];
    }

    // ---- stage A = [input_rep | final_rep] rows as bf16 into LDS ----
    {
        unsigned int* sA = (unsigned int*)smem;
        for (int c = tid; c < TILE_M * 96; c += 256) {   // 96 float4 chunks per row
            int row = c / 96;
            int q   = c - row * 96;
            int r   = row0 + row;
            if (r >= n_rows) r = n_rows - 1;             // clamp; garbage rows skipped later
            float4 v;
            if (q < 32) v = *(const float4*)(input_rep + (size_t)r * MOL_C + q * 4);
            else        v = *(const float4*)(final_rep + (size_t)r * HID_C + (q - 32) * 4);
            uint2 pkd;
            pkd.x = pack_bf16(v.x, v.y);
            pkd.y = pack_bf16(v.z, v.w);
            *(uint2*)(sA + (unsigned)row * (A_STRIDE / 2) + (unsigned)q * 2) = pkd;
        }
    }
    __syncthreads();

    // ---- dual GEMM: acc1 = cat @ W_lin.T ; acc2 = final @ W_last.T ----
    v8f acc1[NS][2], acc2[NS][2];
    {
        v8f vz = {0.f, 0.f, 0.f, 0.f, 0.f, 0.f, 0.f, 0.f};
#pragma unroll
        for (int s = 0; s < NS; ++s) {
            acc1[s][0] = vz; acc1[s][1] = vz;
            acc2[s][0] = vz; acc2[s][1] = vz;
        }
    }

    const int n0 = wave * 32;  // this wave's 32-column slice
    // f32 weight row pointers (fallback path)
    const float* wl0 = W_lin  + (size_t)(n0 +      nlo) * CAT_C;
    const float* wl1 = W_lin  + (size_t)(n0 + 16 + nlo) * CAT_C;
    const float* wt0 = W_last + (size_t)(n0 +      nlo) * HID_C;
    const float* wt1 = W_last + (size_t)(n0 + 16 + nlo) * HID_C;
    // bf16 weight row pointers (fast path); W_last starts at HID_C*CAT_C elems
    const unsigned short* bl0 = wbf + (size_t)(n0 +      nlo) * CAT_C;
    const unsigned short* bl1 = wbf + (size_t)(n0 + 16 + nlo) * CAT_C;
    const unsigned short* bt0 = wbf + (size_t)HID_C * CAT_C + (size_t)(n0 +      nlo) * HID_C;
    const unsigned short* bt1 = wbf + (size_t)HID_C * CAT_C + (size_t)(n0 + 16 + nlo) * HID_C;

    // k in [0,128): only GEMM1 consumes the input_rep part of cat
    for (int k = 0; k < MOL_C; k += 32) {
        v16bf b0, b1;
        if constexpr (WBF) { b0 = load_b_tile_bf(bl0, k, half);  b1 = load_b_tile_bf(bl1, k, half); }
        else               { b0 = load_b_tile_f32(wl0, k, half); b1 = load_b_tile_f32(wl1, k, half); }
        v16bf a = load_a_tile(smem, nlo, k, half);       // strip 0 prefetch
#pragma unroll
        for (int s = 0; s < NS; ++s) {
            v16bf an = (s < NS - 1) ? load_a_tile(smem, (s + 1) * 16 + nlo, k, half) : a;
            acc1[s][0] = WMMA_BF16(a, b0, acc1[s][0]);
            acc1[s][1] = WMMA_BF16(a, b1, acc1[s][1]);
            a = an;
        }
    }
    // k in [128,384): cat[:,128:] == final_rep -> one A load feeds both GEMMs
    for (int k = MOL_C; k < CAT_C; k += 32) {
        v16bf b0, b1, c0, c1;
        if constexpr (WBF) {
            b0 = load_b_tile_bf(bl0, k, half);           b1 = load_b_tile_bf(bl1, k, half);
            c0 = load_b_tile_bf(bt0, k - MOL_C, half);   c1 = load_b_tile_bf(bt1, k - MOL_C, half);
        } else {
            b0 = load_b_tile_f32(wl0, k, half);          b1 = load_b_tile_f32(wl1, k, half);
            c0 = load_b_tile_f32(wt0, k - MOL_C, half);  c1 = load_b_tile_f32(wt1, k - MOL_C, half);
        }
        v16bf a = load_a_tile(smem, nlo, k, half);       // strip 0 prefetch
#pragma unroll
        for (int s = 0; s < NS; ++s) {
            v16bf an = (s < NS - 1) ? load_a_tile(smem, (s + 1) * 16 + nlo, k, half) : a;
            acc1[s][0] = WMMA_BF16(a, b0, acc1[s][0]);
            acc1[s][1] = WMMA_BF16(a, b1, acc1[s][1]);
            acc2[s][0] = WMMA_BF16(a, c0, acc2[s][0]);
            acc2[s][1] = WMMA_BF16(a, c1, acc2[s][1]);
            a = an;
        }
    }

    // ---- epilogue: g = sigmoid(acc1 + b_lin) * (acc2 + b_last) -> LDS ----
    __syncthreads();                       // done reading A tile; reuse LDS for g
    float* gf = (float*)smem;
#pragma unroll
    for (int t = 0; t < 2; ++t) {
        const int col = n0 + t * 16 + nlo; // C/D layout: lane 0..15 == column
        const float bb1 = b_lin[col];
        const float bb2 = b_last[col];
#pragma unroll
        for (int s = 0; s < NS; ++s) {
            const int rbase = s * 16 + half * 8;   // lanes 16..31 hold rows M+8
#pragma unroll
            for (int r = 0; r < 8; ++r) {
                float y1  = acc1[s][t][r] + bb1;
                float att = 1.0f / (1.0f + __expf(-y1));
                float val = att * (acc2[s][t][r] + bb2);
                gf[(unsigned)(rbase + r) * G_STRIDE + (unsigned)col] = val;
            }
        }
    }
    __syncthreads();

    // ---- sorted segment-sum: one thread per column, run-length reduce, ----
    // ---- one f32 global atomic per (graph boundary, column)            ----
    {
        int valid = n_rows - row0;
        if (valid > TILE_M) valid = TILE_M;
        const int col = tid;               // 256 threads == 256 columns
        float sum = 0.0f;
        int cur = s_gi[0];
        for (int rr = 0; rr < valid; ++rr) {
            int gidx = s_gi[rr];
            if (gidx != cur) {
                __hip_atomic_fetch_add(&out[(size_t)cur * HID_C + col], sum,
                                       __ATOMIC_RELAXED, __HIP_MEMORY_SCOPE_AGENT);
                sum = 0.0f;
                cur = gidx;
            }
            sum += gf[(unsigned)rr * G_STRIDE + (unsigned)col];
        }
        __hip_atomic_fetch_add(&out[(size_t)cur * HID_C + col], sum,
                               __ATOMIC_RELAXED, __HIP_MEMORY_SCOPE_AGENT);
    }
}

extern "C" void kernel_launch(void* const* d_in, const int* in_sizes, int n_in,
                              void* d_out, int out_size, void* d_ws, size_t ws_size,
                              hipStream_t stream) {
    const float* input_rep   = (const float*)d_in[0];
    const float* final_rep   = (const float*)d_in[1];
    const int*   graph_index = (const int*)d_in[2];
    const float* W_lin       = (const float*)d_in[3];
    const float* b_lin       = (const float*)d_in[4];
    const float* W_last      = (const float*)d_in[5];
    const float* b_last      = (const float*)d_in[6];
    float* out = (float*)d_out;
    const int n_rows = in_sizes[2];   // N (graph_index length)

    zero_out_kernel<<<(out_size + 255) / 256, 256, 0, stream>>>(out, out_size);

    const int n_wlin  = HID_C * CAT_C;   // 98304
    const int n_wlast = HID_C * HID_C;   // 65536
    const size_t wbf_bytes = (size_t)(n_wlin + n_wlast) * sizeof(unsigned short); // 320 KB
    const int nblocks = (n_rows + TILE_M - 1) / TILE_M;

    if (ws_size >= wbf_bytes) {
        const int npairs = (n_wlin + n_wlast) / 2;
        convert_w_kernel<<<(npairs + 255) / 256, 256, 0, stream>>>(
            W_lin, n_wlin, W_last, n_wlast, (unsigned int*)d_ws);
        attn_pool_kernel<true><<<nblocks, 256, 0, stream>>>(
            input_rep, final_rep, graph_index, W_lin, b_lin, W_last, b_last,
            (const unsigned short*)d_ws, out, n_rows);
    } else {
        attn_pool_kernel<false><<<nblocks, 256, 0, stream>>>(
            input_rep, final_rep, graph_index, W_lin, b_lin, W_last, b_last,
            nullptr, out, n_rows);
    }
}
